// Network_60739427501083
// MI455X (gfx1250) — matrix operand order repaired and checked by reading, not compiled, not run on hardware
//
#include <hip/hip_runtime.h>

// ---------------- problem constants ----------------
#define B_SZ    128
#define C_IN    2312          // 34*34*2
#define HID     512
#define C_OUT   10
#define T_STEPS 300
#define T_PAD   320           // 10 blocks of 32
#define K1_PAD  2336          // 73 * 32
#define K2      512

// SLAYER constants (ts=1, tau=1)
#define DEC    0.36787944117144233f   // exp(-1)
#define CE     2.7182818284590452f    // e
#define CREF  -54.365636569180905f    // -scaleRef*theta*e
#define THETA  10.0f

typedef __attribute__((ext_vector_type(16))) _Float16 v16h;
typedef __attribute__((ext_vector_type(8)))  _Float16 v8h;
typedef __attribute__((ext_vector_type(8)))  float    v8f;
typedef __attribute__((ext_vector_type(4)))  float    v4f;

#define WMMA16(a, b, c) __builtin_amdgcn_wmma_f32_16x16x32_f16( \
        false, (a), false, (b), (short)0, (c), false, false)

// ---------------- workspace layout (bytes) ----------------
// w1h : HID*K1_PAD f16          =   2,392,064
// w2h : 16*K2 f16               =      16,384
// xt  : B*T_PAD*K1_PAD f16      = 191,365,120  (b, t, c) transposed spikes
// u1t : B*T_PAD*HID f32         =  83,886,080  (b, t, o)
// s1h : aliases xt   (xt dead after gemm1)     (b, t, o) f16 = 41,943,040
// u2t : aliases xt + s1h size                  (b, t, 16) f32 = 2,621,440
#define OFF_W1H 0
#define OFF_W2H (OFF_W1H + (size_t)HID * K1_PAD * 2)
#define OFF_XT  (OFF_W2H + (size_t)16 * K2 * 2)
#define OFF_U1T (OFF_XT + (size_t)B_SZ * T_PAD * K1_PAD * 2)
#define OFF_S1H OFF_XT
#define OFF_U2T (OFF_S1H + (size_t)B_SZ * T_PAD * HID * 2)

// ---------------- weight conversion ----------------
__global__ void conv_w1(const float* __restrict__ w1, _Float16* __restrict__ w1h) {
    int idx = blockIdx.x * blockDim.x + threadIdx.x;        // HID*K1_PAD threads
    int o = idx / K1_PAD, k = idx % K1_PAD;
    float v = (k < C_IN) ? w1[(size_t)o * C_IN + k] : 0.0f;
    w1h[idx] = (_Float16)v;
}

__global__ void conv_w2(const float* __restrict__ w2, _Float16* __restrict__ w2h) {
    int idx = blockIdx.x * blockDim.x + threadIdx.x;        // 16*K2 threads
    int o = idx / K2, k = idx % K2;
    float v = (o < C_OUT) ? w2[(size_t)o * K2 + k] : 0.0f;
    w2h[idx] = (_Float16)v;
}

// ---------------- spike transpose+convert: X(b,c,t) f32 -> Xt(b,t,c) f16 ----------------
// 32x32 tiles through LDS; grid (T_PAD/32, K1_PAD/32, B), 256 threads.
__global__ __launch_bounds__(256) void xpose(const float* __restrict__ X,
                                             _Float16* __restrict__ Xt) {
    __shared__ _Float16 lds[32][33];
    const int b  = blockIdx.z;
    const int c0 = blockIdx.y * 32, t0 = blockIdx.x * 32;
    const int tid = threadIdx.x;
    const int tl = tid & 31, cl = tid >> 5;      // read: coalesced along t
    const float* Xb = X + (size_t)b * C_IN * T_STEPS;
    #pragma unroll
    for (int r = 0; r < 4; ++r) {
        int c = c0 + cl + r * 8, t = t0 + tl;
        int cc = (c < C_IN) ? c : (C_IN - 1);          // clamp (branch-free)
        int tc = (t < T_STEPS) ? t : (T_STEPS - 1);
        float v = Xb[(size_t)cc * T_STEPS + tc];
        v = (c < C_IN && t < T_STEPS) ? v : 0.0f;
        lds[cl + r * 8][tl] = (_Float16)v;
    }
    __syncthreads();
    _Float16* Ob = Xt + ((size_t)b * T_PAD + t0) * K1_PAD + c0;
    const int cw = tid & 31, tw = tid >> 5;      // write: coalesced along c
    #pragma unroll
    for (int r = 0; r < 4; ++r)
        Ob[(size_t)(tw + r * 8) * K1_PAD + cw] = lds[cw][tw + r * 8];
}

// ---------------- GEMM1: u1_raw[b,t,o] = sum_c w1[o,c] * Xt[b,t,c] ----------------
// Wave computes a 32(M) x 32(N) block: 4 WMMA per K-step sharing A/B fragments.
// 160 wave-blocks per b (16 M2 x 10 N2); 8 waves per WG -> grid (20, 128).
__global__ __launch_bounds__(256) void gemm1_wmma(
    const _Float16* __restrict__ Xt,    // (B, T_PAD, K1_PAD)
    const _Float16* __restrict__ W,     // (HID, K1_PAD)
    float* __restrict__ U)              // (B, T_PAD, HID)
{
    const int b     = blockIdx.y;
    const int wflat = blockIdx.x * 8 + (threadIdx.x >> 5);  // 0..159
    const int m0    = (wflat & 15) * 32;
    const int n0    = (wflat >> 4) * 32;
    const int lane  = threadIdx.x & 31;
    const int lm    = lane & 15;
    const int kha   = (lane < 16) ? 0 : 8;    // A: K = {kha..+7, kha+16..+23}
    const int khb   = (lane < 16) ? 0 : 8;    // B: K = {khb..+7, khb+16..+23}

    const _Float16* ap = W  + (size_t)(m0 + lm) * K1_PAD + kha;
    const _Float16* bp = Xt + ((size_t)b * T_PAD + n0 + lm) * K1_PAD + khb;
    const size_t AR = (size_t)16 * K1_PAD;    // next M-subtile
    const size_t BR = (size_t)16 * K1_PAD;    // next N-subtile

    v8f c00 = {}, c01 = {}, c10 = {}, c11 = {};
    for (int k0 = 0; k0 < K1_PAD; k0 += 32) {
        v16h a0, a1, b0, b1;
        *(v8h*)&a0       = *(const v8h*)(ap);
        *((v8h*)&a0 + 1) = *(const v8h*)(ap + 16);
        *(v8h*)&a1       = *(const v8h*)(ap + AR);
        *((v8h*)&a1 + 1) = *(const v8h*)(ap + AR + 16);
        *(v8h*)&b0       = *(const v8h*)(bp);
        *((v8h*)&b0 + 1) = *(const v8h*)(bp + 16);
        *(v8h*)&b1       = *(const v8h*)(bp + BR);
        *((v8h*)&b1 + 1) = *(const v8h*)(bp + BR + 16);
        __builtin_prefetch(bp + 512, 0, 1);   // stream-ahead on the HBM-resident operand
        c00 = WMMA16(a0, b0, c00);
        c01 = WMMA16(a0, b1, c01);
        c10 = WMMA16(a1, b0, c10);
        c11 = WMMA16(a1, b1, c11);
        ap += 32; bp += 32;
    }

    // store: element (M=r+8*hi, N=lm) of each 16x16 tile -> (b, t, o); 8 floats
    // per lane are contiguous along o, packed as two float4 stores per tile.
    float* Ub = U + ((size_t)b * T_PAD) * HID;
    const int mhi = (lane >> 4) * 8;
    v8f* accs[4] = { &c00, &c01, &c10, &c11 };
    #pragma unroll
    for (int ti = 0; ti < 4; ++ti) {
        int nn = n0 + ((ti & 1) ? 16 : 0) + lm;
        int mm = m0 + ((ti & 2) ? 16 : 0) + mhi;
        v8f& a = *accs[ti];
        v4f lo = { a[0], a[1], a[2], a[3] };
        v4f hi = { a[4], a[5], a[6], a[7] };
        *(v4f*)&Ub[(size_t)nn * HID + mm]     = lo;
        *(v4f*)&Ub[(size_t)nn * HID + mm + 4] = hi;
    }
}

// ---------------- layer1 fused PSP filter + LIF scan ----------------
__global__ void scan1(const float* __restrict__ U1, _Float16* __restrict__ S1) {
    int tid = blockIdx.x * blockDim.x + threadIdx.x;   // B*HID threads
    int b = tid >> 9, o = tid & (HID - 1);
    const float* u = U1 + (size_t)b * T_PAD * HID + o;
    _Float16*    s = S1 + (size_t)b * T_PAD * HID + o;
    float p = 0.f, q = 0.f, pr = 0.f, qr = 0.f;
    for (int t = 0; t < T_STEPS; ++t) {
        float x = u[(size_t)t * HID];
        q = DEC * (q + p);  p = DEC * p + x;          // PSP alpha filter
        float y = CE * q;
        qr = DEC * (qr + pr);                         // refractory response
        float mem = y + CREF * qr;
        float sp = (mem >= THETA) ? 1.0f : 0.0f;
        pr = DEC * pr + sp;
        s[(size_t)t * HID] = (_Float16)sp;
    }
    for (int t = T_STEPS; t < T_PAD; ++t)
        s[(size_t)t * HID] = (_Float16)0.0f;
}

// ---------------- GEMM2: u2_raw[b,t,o2] = sum_k w2[o2,k] * s1[b,t,k] ----------------
// Wave computes 16(M) x 32(N): 2 WMMA per K-step sharing A. grid (10, 128), 32 thr.
__global__ __launch_bounds__(32) void gemm2_wmma(
    const _Float16* __restrict__ S1,    // (B, T_PAD, HID)
    const _Float16* __restrict__ W2h,   // (16, K2)
    float* __restrict__ U2)             // (B, T_PAD, 16)
{
    const int n0   = blockIdx.x * 32;
    const int b    = blockIdx.y;
    const int lane = threadIdx.x & 31;
    const int lm   = lane & 15;
    const int kha  = (lane < 16) ? 0 : 8;
    const int khb  = (lane < 16) ? 0 : 8;

    const _Float16* ap = W2h + (size_t)lm * K2 + kha;
    const _Float16* bp = S1 + ((size_t)b * T_PAD + n0 + lm) * K2 + khb;
    const size_t BR = (size_t)16 * K2;

    v8f c0 = {}, c1 = {};
    #pragma unroll 4
    for (int k0 = 0; k0 < K2; k0 += 32) {
        v16h a, b0, b1;
        *(v8h*)&a        = *(const v8h*)(ap);
        *((v8h*)&a + 1)  = *(const v8h*)(ap + 16);
        *(v8h*)&b0       = *(const v8h*)(bp);
        *((v8h*)&b0 + 1) = *(const v8h*)(bp + 16);
        *(v8h*)&b1       = *(const v8h*)(bp + BR);
        *((v8h*)&b1 + 1) = *(const v8h*)(bp + BR + 16);
        c0 = WMMA16(a, b0, c0);
        c1 = WMMA16(a, b1, c1);
        ap += 32; bp += 32;
    }
    float* Ub = U2 + ((size_t)b * T_PAD + n0) * 16;
    const int mhi = (lane >> 4) * 8;
    v4f l0 = { c0[0], c0[1], c0[2], c0[3] }, h0 = { c0[4], c0[5], c0[6], c0[7] };
    v4f l1 = { c1[0], c1[1], c1[2], c1[3] }, h1 = { c1[4], c1[5], c1[6], c1[7] };
    *(v4f*)&Ub[(size_t)lm * 16 + mhi]              = l0;
    *(v4f*)&Ub[(size_t)lm * 16 + mhi + 4]          = h0;
    *(v4f*)&Ub[(size_t)(lm + 16) * 16 + mhi]       = l1;
    *(v4f*)&Ub[(size_t)(lm + 16) * 16 + mhi + 4]   = h1;
}

// ---------------- layer2 fused PSP filter + LIF scan -> final output ----------------
__global__ void scan2(const float* __restrict__ U2, float* __restrict__ out) {
    int tid = blockIdx.x * blockDim.x + threadIdx.x;   // B*16 threads
    if (tid >= B_SZ * 16) return;
    int b = tid >> 4, o = tid & 15;
    if (o >= C_OUT) return;
    const float* u = U2 + (size_t)b * T_PAD * 16 + o;
    float* so = out + ((size_t)b * C_OUT + o) * T_STEPS;
    float p = 0.f, q = 0.f, pr = 0.f, qr = 0.f;
    for (int t = 0; t < T_STEPS; ++t) {
        float x = u[(size_t)t * 16];
        q = DEC * (q + p);  p = DEC * p + x;
        float y = CE * q;
        qr = DEC * (qr + pr);
        float mem = y + CREF * qr;
        float sp = (mem >= THETA) ? 1.0f : 0.0f;
        pr = DEC * pr + sp;
        so[t] = sp;
    }
}

extern "C" void kernel_launch(void* const* d_in, const int* in_sizes, int n_in,
                              void* d_out, int out_size, void* d_ws, size_t ws_size,
                              hipStream_t stream) {
    const float* X  = (const float*)d_in[0];   // (128, 2312, 300)
    const float* W1 = (const float*)d_in[1];   // (512, 2312)
    const float* W2 = (const float*)d_in[2];   // (10, 512)
    float* out = (float*)d_out;                // (128, 10, 300)

    char* ws = (char*)d_ws;
    _Float16* w1h = (_Float16*)(ws + OFF_W1H);
    _Float16* w2h = (_Float16*)(ws + OFF_W2H);
    _Float16* xt  = (_Float16*)(ws + OFF_XT);
    float*    u1t = (float*)   (ws + OFF_U1T);
    _Float16* s1h = (_Float16*)(ws + OFF_S1H);   // aliases xt (dead by then)
    float*    u2t = (float*)   (ws + OFF_U2T);

    conv_w1<<<(HID * K1_PAD) / 256, 256, 0, stream>>>(W1, w1h);
    conv_w2<<<(16 * K2) / 256,      256, 0, stream>>>(W2, w2h);

    // spikes -> transposed padded f16 (streaming pre-pass)
    xpose<<<dim3(T_PAD / 32, K1_PAD / 32, B_SZ), 256, 0, stream>>>(X, xt);

    // layer 1: time-batched GEMM (alpha filter commuted past matmul) + fused scan
    gemm1_wmma<<<dim3(20, B_SZ), 256, 0, stream>>>(xt, w1h, u1t);
    scan1<<<(B_SZ * HID) / 256, 256, 0, stream>>>(u1t, s1h);

    // layer 2: small GEMM + fused scan writing the final spike train
    gemm2_wmma<<<dim3(T_PAD / 32, B_SZ), 32, 0, stream>>>(s1h, w2h, u2t);
    scan2<<<(B_SZ * 16 + 255) / 256, 256, 0, stream>>>(u2t, out);
}
